// MHA_24008867184862
// MI455X (gfx1250) — matrix operand order repaired and checked by reading, not compiled, not run on hardware
//
#include <hip/hip_runtime.h>

// ---------------------------------------------------------------------------
// MI455X (gfx1250) fused MHA pipeline, bf16 WMMA + async-to-LDS everywhere.
//   P1: convert x -> bf16 [M][K]
//   P2: convert+transpose Wqkv, Wout -> bf16 [N][K]
//   K1: qkv = x @ Wqkv + bqkv      (async-to-LDS double buffer + wmma bf16)
//   K2: RoPE + convert -> bf16 q [S][2048], bf16 k [S][512]
//   K2b: transpose v -> bf16 V^T [512][S]
//   K3: flash-style causal GQA attention (async-to-LDS K/V staging, wmma bf16)
//   K4: out = ctx @ Wout + bout    (same GEMM)
// ---------------------------------------------------------------------------

typedef __attribute__((ext_vector_type(16))) __bf16        v16bf;
typedef __attribute__((ext_vector_type(8)))  float         v8f;
typedef __attribute__((ext_vector_type(4)))  float         fl4;
typedef __attribute__((ext_vector_type(4)))  unsigned int  ui4;
typedef __attribute__((ext_vector_type(2)))  unsigned int  ui2;

__device__ __forceinline__ unsigned f2bf_bits(float f) {
  union { float f; unsigned u; } c; c.f = f;
  unsigned r = c.u + 0x7FFFu + ((c.u >> 16) & 1u);   // round-to-nearest-even
  return r >> 16;
}
__device__ __forceinline__ unsigned packbf(float lo, float hi) {
  return (f2bf_bits(hi) << 16) | f2bf_bits(lo);
}

// 16x32 bf16 WMMA operand fragment (8 VGPRs) built from raw bf16 bit patterns.
union Frag {
  v16bf v;
  unsigned u[8];
  ui4 q[2];
};

__device__ __forceinline__ float rmax16(float v) {
  for (int o = 8; o > 0; o >>= 1) v = fmaxf(v, __shfl_xor(v, o, 16));
  return v;
}
__device__ __forceinline__ float rsum16(float v) {
  for (int o = 8; o > 0; o >>= 1) v += __shfl_xor(v, o, 16);
  return v;
}

// LDS byte address of a shared object = low 32 bits of its flat address
// (flat shared = {aperture[63:32], lds_offset[31:0]} per CDNA5 aperture rules).
__device__ __forceinline__ unsigned lds_addr(const void* p) {
  return (unsigned)(unsigned long long)p;
}

// Async global->LDS copy of 16 bytes, GVS mode (SGPR base + VGPR byte offset).
__device__ __forceinline__ void async_copy16(unsigned lds_off, const void* base,
                                             unsigned byte_off) {
  asm volatile("global_load_async_to_lds_b128 %0, %1, %2"
               :: "v"(lds_off), "v"(byte_off), "s"((unsigned long long)base)
               : "memory");
}

// ---------------------------------------------------------------------------
// P1: elementwise f32 -> bf16
// ---------------------------------------------------------------------------
__global__ void convert_bf16(const float* __restrict__ in,
                             unsigned short* __restrict__ out, int n)
{
  int i = (blockIdx.x * blockDim.x + threadIdx.x) * 4;
  if (i >= n) return;
  fl4 v = *reinterpret_cast<const fl4*>(in + i);
  ui2 p; p[0] = packbf(v[0], v[1]); p[1] = packbf(v[2], v[3]);
  *reinterpret_cast<ui2*>(out + i) = p;
}

// ---------------------------------------------------------------------------
// P2: W [K][N] f32 (row stride ldw)  ->  Wt [N][K] bf16 (32x32 LDS transpose)
// ---------------------------------------------------------------------------
__global__ __launch_bounds__(256) void transpose_to_bf16(
    const float* __restrict__ W, unsigned short* __restrict__ Wt,
    int K, int N, int ldw)
{
  __shared__ unsigned short t[32][33];
  const int k0 = blockIdx.y * 32, n0 = blockIdx.x * 32;
  const int tx = threadIdx.x & 31, ty0 = threadIdx.x >> 5;
  for (int i = 0; i < 4; ++i) {
    int ky = ty0 + i * 8;
    t[tx][ky] = (unsigned short)f2bf_bits(W[(size_t)(k0 + ky) * ldw + n0 + tx]);
  }
  __syncthreads();
  for (int i = 0; i < 4; ++i) {
    int ny = ty0 + i * 8;
    Wt[(size_t)(n0 + ny) * K + k0 + tx] = t[ny][tx];
  }
}

// ---------------------------------------------------------------------------
// C = A(MxK,bf16) * Wt(NxK,bf16)^T + bias : 128x128x32 tiles.
// 8 waves; wave grid 4(M) x 2(N); each wave -> 32x64 output (8 wmma).
// Staging via GLOBAL_LOAD_ASYNC_TO_LDS_B128 into double-buffered LDS.
// ---------------------------------------------------------------------------
__global__ __launch_bounds__(256) void wmma_gemm_async(
    const unsigned short* __restrict__ Ab,   // [M][K] bf16
    const unsigned short* __restrict__ Wt,   // [N][K] bf16
    const float* __restrict__ bias, float* __restrict__ C,
    int M, int N, int K)
{
  __shared__ alignas(16) unsigned short lA[2][128][32];   // [m][k]
  __shared__ alignas(16) unsigned short lB[2][128][32];   // [n][k]

  const int tid  = threadIdx.x;
  const int wave = tid >> 5, lane = tid & 31;
  const int half = lane >> 4, l16 = lane & 15;
  const int bm0 = blockIdx.y * 128, bn0 = blockIdx.x * 128;
  const int mb = (wave >> 1) * 32;
  const int nb = (wave & 1) * 64;

  v8f acc[2][4];
  for (int i = 0; i < 2; ++i)
    for (int j = 0; j < 4; ++j)
      for (int r = 0; r < 8; ++r) acc[i][j][r] = 0.0f;

  const int nkt = K >> 5;

  // issue one 128x32 bf16 tile (A and B) as 4 async b128 copies per thread
  auto issue_tile = [&](int kt, int buf) {
    for (int it = 0; it < 2; ++it) {
      int c = tid + it * 256;               // 0..511 chunks of 16B
      int row = c >> 2, seg = c & 3;        // 4 chunks per 64B row
      unsigned aoff = (unsigned)(((size_t)(bm0 + row) * K + kt * 32 + seg * 8) * 2);
      unsigned boff = (unsigned)(((size_t)(bn0 + row) * K + kt * 32 + seg * 8) * 2);
      async_copy16(lds_addr(&lA[buf][row][seg * 8]), Ab, aoff);
      async_copy16(lds_addr(&lB[buf][row][seg * 8]), Wt, boff);
    }
  };

  issue_tile(0, 0);
  for (int kt = 0; kt < nkt; ++kt) {
    const int buf = kt & 1;
    __syncthreads();                         // readers of buf^1 are done
    if (kt + 1 < nkt) {
      issue_tile(kt + 1, buf ^ 1);
      asm volatile("s_wait_asynccnt 0x4" ::: "memory");  // tile kt landed (in-order)
    } else {
      asm volatile("s_wait_asynccnt 0x0" ::: "memory");
    }
    __syncthreads();                         // all waves' tile-kt data visible

    // ---- fragments per ISA 16-bit layouts
    Frag fa[2], fb[4];
    const int akb = half * 8;                // A: lanes 0-15 K 0..7/16..23; 16-31 K 8..15/24..31
    for (int i = 0; i < 2; ++i) {
      const unsigned short* p = &lA[buf][mb + i * 16 + l16][akb];
      fa[i].q[0] = *reinterpret_cast<const ui4*>(p);
      fa[i].q[1] = *reinterpret_cast<const ui4*>(p + 16);
    }
    const int bkb = half * 8;                // B: lanes 0-15 K 0..7/16..23; 16-31 K 8..15/24..31
    for (int j = 0; j < 4; ++j) {
      const unsigned short* p = &lB[buf][nb + j * 16 + l16][bkb];
      fb[j].q[0] = *reinterpret_cast<const ui4*>(p);
      fb[j].q[1] = *reinterpret_cast<const ui4*>(p + 16);
    }
    for (int i = 0; i < 2; ++i)
      for (int j = 0; j < 4; ++j)
        acc[i][j] = __builtin_amdgcn_wmma_f32_16x16x32_bf16(
            false, fa[i].v, false, fb[j].v, (short)0, acc[i][j], false, false);
  }

  // ---- epilogue: bias add, fp32 store
  for (int j = 0; j < 4; ++j) {
    int gn = bn0 + nb + j * 16 + l16;
    float bs = bias[gn];
    for (int i = 0; i < 2; ++i)
      for (int r = 0; r < 8; ++r) {
        int gm = bm0 + mb + i * 16 + half * 8 + r;
        C[(size_t)gm * N + gn] = acc[i][j][r] + bs;
      }
  }
}

// ---------------------------------------------------------------------------
// RoPE + precision convert. Reads f32 qkv rows, writes:
//   qb [S][2048] bf16 (32 heads, rotary dims rotated, rest converted)
//   kb [S][512]  bf16 (8 kv heads, same treatment)
// qkv row layout: [0:2048]=q, [2048:2560]=k, [2560:3072]=v
// ---------------------------------------------------------------------------
__global__ void rope_convert(const float* __restrict__ qkv,
                             unsigned short* __restrict__ qb,
                             unsigned short* __restrict__ kb, int S)
{
  int idx = blockIdx.x * blockDim.x + threadIdx.x;   // S * 40 * 16 threads
  int d = idx & 15;
  int t = idx >> 4;
  int hh = t % 40;
  int s = t / 40;
  if (s >= S) return;
  const float* p = qkv + (size_t)s * 3072 + ((hh < 32) ? hh * 64 : 2048 + (hh - 32) * 64);
  float invf = powf(10000.0f, -(float)d * (1.0f / 16.0f));
  float fr = (float)s * invf;
  float c = cosf(fr), sn = sinf(fr);
  float x1 = p[d], x2 = p[d + 16];
  unsigned short o1 = (unsigned short)f2bf_bits(x1 * c - x2 * sn);
  unsigned short o2 = (unsigned short)f2bf_bits(x1 * sn + x2 * c);
  unsigned short o3 = (unsigned short)f2bf_bits(p[d + 32]);   // pass-through dims
  unsigned short o4 = (unsigned short)f2bf_bits(p[d + 48]);
  unsigned short* o = (hh < 32) ? (qb + (size_t)s * 2048 + hh * 64)
                                : (kb + (size_t)s * 512 + (hh - 32) * 64);
  o[d] = o1; o[d + 16] = o2; o[d + 32] = o3; o[d + 48] = o4;
}

// ---------------------------------------------------------------------------
// Flash-style causal attention, GQA 4:1. Grid: (S/128, 32 heads).
// 8 waves/block, each wave owns a 16-row q tile. K and V^T tiles staged via
// async-to-LDS (double buffered); online softmax fp32; bf16 ctx output.
// ---------------------------------------------------------------------------
__global__ __launch_bounds__(256) void attn_fa(
    const unsigned short* __restrict__ qb,    // [S][2048] bf16 roped q
    const unsigned short* __restrict__ kb,    // [S][512]  bf16 roped k
    const unsigned short* __restrict__ vt,    // [512][S]  bf16 V^T
    unsigned short* __restrict__ ctxb, int S)
{
  __shared__ alignas(16) unsigned short lK[2][32][64];   // [key][d]  (B-frag q@K^T)
  __shared__ alignas(16) unsigned short lV[2][64][32];   // [d][key]  (B-frag P@V)
  __shared__ alignas(16) unsigned short lP[8][16][32];   // per-wave P relayout bounce

  const int tid = threadIdx.x, wave = tid >> 5, lane = tid & 31;
  const int half = lane >> 4, l16 = lane & 15;
  const int h = blockIdx.y, kvh = h >> 2;
  const int qrow0 = blockIdx.x * 128;
  const int m0 = qrow0 + wave * 16;
  const float scale = 0.125f;                            // 1/sqrt(64)

  // issue one 32-key tile: K rows (32 x 128B) + V^T rows (64 x 64B),
  // 512 16B chunks total -> 2 async b128 copies per thread
  auto issue_tile = [&](int kt, int buf) {
    int krow = tid >> 3, kseg = tid & 7;                 // 8 chunks per K row
    unsigned koff = (unsigned)((((size_t)(kt * 32 + krow) * 512) + kvh * 64 + kseg * 8) * 2);
    async_copy16(lds_addr(&lK[buf][krow][kseg * 8]), kb, koff);
    int vd = tid >> 2, vseg = tid & 3;                   // 4 chunks per V^T row
    unsigned voff = (unsigned)((((size_t)(kvh * 64 + vd) * S) + kt * 32 + vseg * 8) * 2);
    async_copy16(lds_addr(&lV[buf][vd][vseg * 8]), vt, voff);
  };

  // q A-fragments (d 0..31, 32..63) straight from bf16 buffer in A layout
  Frag qa[2];
  {
    const unsigned short* qp = qb + (size_t)(m0 + l16) * 2048 + h * 64;
    for (int f = 0; f < 2; ++f) {
      const unsigned short* p = qp + f * 32 + half * 8;
      qa[f].q[0] = *reinterpret_cast<const ui4*>(p);
      qa[f].q[1] = *reinterpret_cast<const ui4*>(p + 16);
    }
  }

  float mrow[8], lrow[8];
  v8f oacc[4];
  for (int r = 0; r < 8; ++r) { mrow[r] = -3.0e38f; lrow[r] = 0.0f; }
  for (int j = 0; j < 4; ++j)
    for (int r = 0; r < 8; ++r) oacc[j][r] = 0.0f;

  const int nkt = (qrow0 >> 5) + 4;   // causal span of this 128-row block
  issue_tile(0, 0);
  for (int kt = 0; kt < nkt; ++kt) {
    const int buf = kt & 1;
    __syncthreads();                         // readers of buf^1 are done
    if (kt + 1 < nkt) {
      issue_tile(kt + 1, buf ^ 1);
      asm volatile("s_wait_asynccnt 0x2" ::: "memory");  // tile kt landed
    } else {
      asm volatile("s_wait_asynccnt 0x0" ::: "memory");
    }
    __syncthreads();                         // all waves' tile-kt data visible
    if (kt * 32 > m0 + 15) continue;         // wave's rows not reached (syncs stay uniform)

    // ---- scores S(16x32) = q @ K^T : two 16x16 accumulators, K-dim 64 = 2 wmma
    v8f sc[2];
    for (int j = 0; j < 2; ++j)
      for (int r = 0; r < 8; ++r) sc[j][r] = 0.0f;
    for (int j = 0; j < 2; ++j)
      for (int f = 0; f < 2; ++f) {
        Frag kf;
        const unsigned short* p = &lK[buf][j * 16 + l16][f * 32 + half * 8];
        kf.q[0] = *reinterpret_cast<const ui4*>(p);
        kf.q[1] = *reinterpret_cast<const ui4*>(p + 16);
        sc[j] = __builtin_amdgcn_wmma_f32_16x16x32_bf16(
            false, qa[f].v, false, kf.v, (short)0, sc[j], false, false);
      }

    // ---- scale + causal mask
    for (int j = 0; j < 2; ++j) {
      int col = kt * 32 + j * 16 + l16;
      for (int r = 0; r < 8; ++r) {
        int row = m0 + half * 8 + r;
        float s = sc[j][r] * scale;
        sc[j][r] = (col > row) ? -3.0e38f : s;
      }
    }

    // ---- online softmax update (per-row stats via 16-lane xor reductions)
    for (int r = 0; r < 8; ++r) {
      float rm   = rmax16(fmaxf(sc[0][r], sc[1][r]));
      float mnew = fmaxf(mrow[r], rm);
      float corr = __expf(mrow[r] - mnew);
      float p0 = __expf(sc[0][r] - mnew);
      float p1 = __expf(sc[1][r] - mnew);
      sc[0][r] = p0; sc[1][r] = p1;
      float rs = rsum16(p0 + p1);
      lrow[r] = lrow[r] * corr + rs;
      mrow[r] = mnew;
      for (int j = 0; j < 4; ++j) oacc[j][r] *= corr;
    }

    // ---- P: C-layout -> A-layout via per-wave LDS bounce
    for (int j = 0; j < 2; ++j)
      for (int r = 0; r < 8; ++r)
        lP[wave][half * 8 + r][j * 16 + l16] = (unsigned short)f2bf_bits(sc[j][r]);
    asm volatile("s_wait_dscnt 0x0" ::: "memory");
    Frag pa;
    {
      const unsigned short* p = &lP[wave][l16][half * 8];
      pa.q[0] = *reinterpret_cast<const ui4*>(p);
      pa.q[1] = *reinterpret_cast<const ui4*>(p + 16);
    }

    // ---- out(16x64) += P(16x32) @ V(32x64)
    for (int j = 0; j < 4; ++j) {
      Frag vf;
      const unsigned short* p = &lV[buf][j * 16 + l16][half * 8];
      vf.q[0] = *reinterpret_cast<const ui4*>(p);
      vf.q[1] = *reinterpret_cast<const ui4*>(p + 16);
      oacc[j] = __builtin_amdgcn_wmma_f32_16x16x32_bf16(
          false, pa.v, false, vf.v, (short)0, oacc[j], false, false);
    }
  }

  // ---- normalize and store context [S][2048] as bf16 (feeds GEMM2 directly)
  for (int j = 0; j < 4; ++j)
    for (int r = 0; r < 8; ++r) {
      int row = m0 + half * 8 + r;
      ctxb[(size_t)row * 2048 + h * 64 + j * 16 + l16] =
          (unsigned short)f2bf_bits(oacc[j][r] / lrow[r]);
    }
}

// ---------------------------------------------------------------------------
extern "C" void kernel_launch(void* const* d_in, const int* in_sizes, int n_in,
                              void* d_out, int out_size, void* d_ws, size_t ws_size,
                              hipStream_t stream)
{
  (void)in_sizes; (void)n_in; (void)out_size; (void)ws_size;
  const float* x    = (const float*)d_in[0];   // [1,2048,2048]
  const float* Wqkv = (const float*)d_in[1];   // [2048,3072]
  const float* bqkv = (const float*)d_in[2];   // [3072]
  const float* Wout = (const float*)d_in[3];   // [2048,2048]
  const float* bout = (const float*)d_in[4];   // [2048]
  float* out = (float*)d_out;                  // [1,2048,2048]

  const int S = 2048, D = 2048, QKV = 3072;

  // workspace layout (~60 MB total, regions reused across pipeline phases)
  float* qkv = (float*)d_ws;                                        // S*QKV f32  (24 MB)
  unsigned short* xb    = (unsigned short*)(qkv + (size_t)S * QKV); // S*D bf16   (8 MB)
  unsigned short* wqkvt = xb + (size_t)S * D;                       // QKV*D bf16 (12 MB)
  unsigned short* woutt = wqkvt + (size_t)D * QKV;                  // D*D bf16   (8 MB)
  unsigned short* ctxb  = woutt + (size_t)D * D;                    // S*D bf16   (8 MB)
  // dead-after-GEMM1 regions reused by the attention phase:
  unsigned short* qb = xb;                        // roped q bf16 [S][2048]
  unsigned short* kb = wqkvt;                     // roped k bf16 [S][512]  (2 MB)
  unsigned short* vt = wqkvt + (size_t)S * 512;   // V^T bf16 [512][S]      (2 MB)

  // P1/P2: one-time precision conversion + weight transposes
  convert_bf16<<<(S * D / 4 + 255) / 256, 256, 0, stream>>>(x, xb, S * D);
  transpose_to_bf16<<<dim3(QKV / 32, D / 32), 256, 0, stream>>>(Wqkv, wqkvt, D, QKV, QKV);
  transpose_to_bf16<<<dim3(D / 32, D / 32), 256, 0, stream>>>(Wout, woutt, D, D, D);

  // 1) qkv = x @ Wqkv + bqkv
  wmma_gemm_async<<<dim3(QKV / 128, S / 128), 256, 0, stream>>>(xb, wqkvt, bqkv, qkv, S, QKV, D);
  // 2) RoPE + convert (q,k) to bf16; transpose v to V^T bf16
  rope_convert<<<(S * 40 * 16) / 256, 256, 0, stream>>>(qkv, qb, kb, S);
  transpose_to_bf16<<<dim3(512 / 32, S / 32), 256, 0, stream>>>(qkv + 2560, vt, S, 512, QKV);
  // 3) causal GQA attention -> ctx (bf16)
  attn_fa<<<dim3(S / 128, 32), 256, 0, stream>>>(qb, kb, vt, ctxb, S);
  // 4) out = ctx @ Wout + bout
  wmma_gemm_async<<<dim3(D / 128, S / 128), 256, 0, stream>>>(ctxb, woutt, bout, out, S, D, D);
}
